// LinkPredictionPrompt_5368709120801
// MI455X (gfx1250) — matrix operand order repaired and checked by reading, not compiled, not run on hardware
//
#include <hip/hip_runtime.h>

// ---------------------------------------------------------------------------
// LinkPrediction fused pipeline for MI455X (gfx1250, wave32, WMMA + TDM)
//   h   = x @ W1^T + b1            f32 WMMA 16x16x4
//   h   = BN(h) -> ReLU            two-pass column reduction
//   emb = h @ W2^T + b2            f32 WMMA 16x16x4, split to bf16 hi/lo
//   out = mask(sigmoid(emb@emb^T)) bf16 WMMA 16x16x32 (bf16x3 ~ fp32),
//                                  TDM tensor_load_to_lds panel staging
//   out[src,dst] = 1.0             edge scatter
// ---------------------------------------------------------------------------

typedef __attribute__((ext_vector_type(2)))  float        v2f;
typedef __attribute__((ext_vector_type(8)))  float        v8f;
typedef __attribute__((ext_vector_type(16))) __bf16       v16bf;
typedef __attribute__((ext_vector_type(4)))  unsigned int u32x4;
typedef __attribute__((ext_vector_type(8)))  int          i32x8;
typedef __attribute__((ext_vector_type(4)))  int          i32x4;

union BfOp { v16bf v; uint4 u[2]; };

constexpr int N_NODES = 8192;
constexpr int EMB     = 256;
constexpr int HID     = 512;
constexpr int OUTD    = 128;

// SYRK LDS staging: 128 rows x 128 bf16 per panel, padded 4 dwords per row
// (row stride 272B) so fragment reads spread over all 64 LDS banks.
constexpr int LDS_ROW   = 272;                 // 256B data + 16B pad
constexpr int LDS_PANEL = 128 * LDS_ROW;       // 34816 B
constexpr int LDS_AHI   = 0;
constexpr int LDS_ALO   = LDS_PANEL;
constexpr int LDS_BHI   = 2 * LDS_PANEL;
constexpr int LDS_BLO   = 3 * LDS_PANEL;
constexpr int LDS_TOTAL = 4 * LDS_PANEL;       // 139264 B (< 320KB/WGP)

__device__ __forceinline__ unsigned short f32_to_bf16(float f) {
  unsigned int u = __float_as_uint(f);
  u += 0x7FFFu + ((u >> 16) & 1u);             // round-to-nearest-even
  return (unsigned short)(u >> 16);
}
__device__ __forceinline__ float bf16_to_f32(unsigned short h) {
  return __uint_as_float(((unsigned int)h) << 16);
}

// ---- TDM: async 2D tile load (128 rows x 128 bf16, row stride 256B) -------
// D# built per CDNA5 ISA 8.3-8.5: group0 = {count, lds_addr, global_addr,
// type=2}; group1 = {data_size=2B, pad 4dw/64dw, dims/strides}; groups 2/3
// zero (2D tile).  One instruction moves a full 32KB panel into LDS.
__device__ __forceinline__ void tdm_load_panel_128x128_bf16(
    const unsigned short* gptr, unsigned int lds_byte_off)
{
  unsigned long long ga = (unsigned long long)gptr;
  u32x4 g0;
  g0[0] = 1u;                                        // count=1, user mode
  g0[1] = lds_byte_off;                              // LDS byte address
  g0[2] = (unsigned int)(ga & 0xFFFFFFFFu);          // global_addr[31:0]
  g0[3] = (unsigned int)((ga >> 32) & 0x1FFFFFFu)    // global_addr[56:32]
        | (2u << 30);                                // type = 2 ("image")
  i32x8 g1;
  g1[0] = (int)0x07510000u;    // data_size=1(2B) | pad_en | intvl=64dw | amt=4dw
  g1[1] = (int)(128u << 16);   // tensor_dim0 = 128 (K elems, contiguous)
  g1[2] = (int)(8192u << 16);  // tensor_dim1 = 8192 rows (no OOB clip)
  g1[3] = (int)(128u << 16);   // tile_dim0 = 128
  g1[4] = (int)128u;           // tile_dim1 = 128 rows, tile_dim2 = 0 (2D)
  g1[5] = (int)128u;           // tensor_dim0_stride = 128 elems
  g1[6] = 0;
  g1[7] = 0;
  i32x4 g2 = {};               // unused higher dims
  i32x4 g3 = {};
#if defined(__clang_major__) && (__clang_major__ >= 23)
  i32x8 g4 = {};
  __builtin_amdgcn_tensor_load_to_lds(g0, g1, g2, g3, g4, 0);
#else
  __builtin_amdgcn_tensor_load_to_lds(g0, g1, g2, g3, 0);
#endif
}

// ---------------- GEMM1: h = x @ W1^T + b1  (f32 WMMA 16x16x4) -------------
// one 16x16 tile per wave; tiles = 512 (M) x 32 (N) = 16384 waves
__global__ __launch_bounds__(256) void gemm1_kernel(
    const float* __restrict__ x, const float* __restrict__ W1,
    const float* __restrict__ b1, float* __restrict__ h)
{
  const int lane = threadIdx.x & 31;
  const int wave = blockIdx.x * 8 + (threadIdx.x >> 5);
  const int tm = wave >> 5;        // 0..511
  const int tn = wave & 31;        // 0..31
  const int l15 = lane & 15;
  const int hs  = lane >> 4;

  const float* pa = x  + (size_t)(tm * 16 + l15) * EMB + 2 * hs;
  const float* pb = W1 + (size_t)(tn * 16 + l15) * EMB + 2 * hs;

  v8f acc = {};
  #pragma unroll 8
  for (int k0 = 0; k0 < EMB; k0 += 4) {
    v2f a = *(const v2f*)(pa + k0);
    v2f b = *(const v2f*)(pb + k0);
    acc = __builtin_amdgcn_wmma_f32_16x16x4_f32(
        false, a, false, b, (short)0, acc, false, false);
  }

  const int col  = tn * 16 + l15;
  const float bias = b1[col];
  #pragma unroll
  for (int r = 0; r < 8; ++r) {
    int row = tm * 16 + r + 8 * hs;
    h[(size_t)row * HID + col] = acc[r] + bias;
  }
}

// ---------------- BatchNorm: partial column sums ---------------------------
__global__ __launch_bounds__(512) void bn_partial_kernel(
    const float* __restrict__ h, float* __restrict__ partial)
{
  const int c = threadIdx.x;
  const int r0 = blockIdx.x * 128;
  float s = 0.f, q = 0.f;
  for (int r = r0; r < r0 + 128; ++r) {
    float v = h[(size_t)r * HID + c];
    s += v; q += v * v;
  }
  partial[blockIdx.x * HID + c]            = s;
  partial[64 * HID + blockIdx.x * HID + c] = q;
}

// ---------------- BatchNorm: finalize scale/shift --------------------------
__global__ __launch_bounds__(512) void bn_finalize_kernel(
    const float* __restrict__ partial, const float* __restrict__ gamma,
    const float* __restrict__ beta, float* __restrict__ scale,
    float* __restrict__ shiftv)
{
  const int c = threadIdx.x;
  float s = 0.f, q = 0.f;
  for (int b = 0; b < 64; ++b) {
    s += partial[b * HID + c];
    q += partial[64 * HID + b * HID + c];
  }
  const float inv_n = 1.0f / (float)N_NODES;
  float mu  = s * inv_n;
  float var = q * inv_n - mu * mu;            // biased variance, as BN uses
  float sc  = gamma[c] * rsqrtf(var + 1e-5f);
  scale[c]  = sc;
  shiftv[c] = beta[c] - mu * sc;
}

// ---------------- BatchNorm apply + ReLU (in place) ------------------------
__global__ __launch_bounds__(256) void bn_apply_kernel(
    float* __restrict__ h, const float* __restrict__ scale,
    const float* __restrict__ shiftv)
{
  int i = blockIdx.x * 256 + threadIdx.x;
  if (i < N_NODES * HID) {
    int c = i & (HID - 1);
    float v = h[i] * scale[c] + shiftv[c];
    h[i] = fmaxf(v, 0.0f);
  }
}

// ---------------- GEMM2: emb = hn @ W2^T + b2 -> bf16 hi/lo ----------------
__global__ __launch_bounds__(256) void gemm2_kernel(
    const float* __restrict__ hn, const float* __restrict__ W2,
    const float* __restrict__ b2, unsigned short* __restrict__ ehi,
    unsigned short* __restrict__ elo)
{
  const int lane = threadIdx.x & 31;
  const int wave = blockIdx.x * 8 + (threadIdx.x >> 5);
  const int tm = wave >> 3;        // 0..511
  const int tn = wave & 7;         // 0..7
  const int l15 = lane & 15;
  const int hs  = lane >> 4;

  const float* pa = hn + (size_t)(tm * 16 + l15) * HID + 2 * hs;
  const float* pb = W2 + (size_t)(tn * 16 + l15) * HID + 2 * hs;

  v8f acc = {};
  #pragma unroll 8
  for (int k0 = 0; k0 < HID; k0 += 4) {
    v2f a = *(const v2f*)(pa + k0);
    v2f b = *(const v2f*)(pb + k0);
    acc = __builtin_amdgcn_wmma_f32_16x16x4_f32(
        false, a, false, b, (short)0, acc, false, false);
  }

  const int col  = tn * 16 + l15;
  const float bias = b2[col];
  #pragma unroll
  for (int r = 0; r < 8; ++r) {
    int row = tm * 16 + r + 8 * hs;
    float e = acc[r] + bias;
    unsigned short hi = f32_to_bf16(e);
    unsigned short lo = f32_to_bf16(e - bf16_to_f32(hi));
    ehi[(size_t)row * OUTD + col] = hi;
    elo[(size_t)row * OUTD + col] = lo;
  }
}

// ---------------- SYRK + sigmoid + triu/threshold --------------------------
// 128x128 output per workgroup (8 waves). TDM stages the A/B bf16 hi/lo
// panels into padded LDS; each wave computes a 64x32 sub-tile with
// bf16x3 WMMA (hi*hi + hi*lo + lo*hi). Blocks strictly below the diagonal
// store zeros with no loads or compute.
__global__ __launch_bounds__(256) void syrk_sigmoid_kernel(
    const unsigned short* __restrict__ ehi,
    const unsigned short* __restrict__ elo,
    float* __restrict__ out)
{
  extern __shared__ char smem[];

  const int bm = blockIdx.x >> 6;          // 0..63
  const int bn = blockIdx.x & 63;          // 0..63
  const int mBase = bm * 128;
  const int nBase = bn * 128;

  if (mBase >= nBase + 128) {              // strictly lower-triangular block
    const float4 z4 = make_float4(0.f, 0.f, 0.f, 0.f);
    #pragma unroll
    for (int i = 0; i < 16; ++i) {
      int idx = threadIdx.x + 256 * i;     // 0..4095
      int row = idx >> 5;                  // 0..127
      int c4  = idx & 31;                  // 0..31
      *(float4*)(out + (size_t)(mBase + row) * N_NODES + nBase + 4 * c4) = z4;
    }
    return;
  }

  const int lane = threadIdx.x & 31;
  const int wave = threadIdx.x >> 5;
  const int l15  = lane & 15;
  const int hs   = lane >> 4;
  const int wm   = wave >> 2;              // 0..1 -> 64 rows each
  const int wn   = wave & 3;               // 0..3 -> 32 cols each

  // waves 0..3 each issue one TDM panel load (async, TENSORcnt-tracked)
  if (wave == 0)
    tdm_load_panel_128x128_bf16(ehi + (size_t)mBase * OUTD, LDS_AHI);
  else if (wave == 1)
    tdm_load_panel_128x128_bf16(elo + (size_t)mBase * OUTD, LDS_ALO);
  else if (wave == 2)
    tdm_load_panel_128x128_bf16(ehi + (size_t)nBase * OUTD, LDS_BHI);
  else if (wave == 3)
    tdm_load_panel_128x128_bf16(elo + (size_t)nBase * OUTD, LDS_BLO);
  if (wave < 4) __builtin_amdgcn_s_wait_tensorcnt(0);
  __syncthreads();

  v8f acc[4][2] = {};
  const int aK = 8 * hs;    // A frag: lane-half -> K {0-7,16-23}/{8-15,24-31}
  const int bK = 8 * hs;    // B frag: lane-half -> K {0-7,16-23}/{8-15,24-31}

  for (int k0 = 0; k0 < OUTD; k0 += 32) {
    BfOp aH[4], aL[4];
    #pragma unroll
    for (int m = 0; m < 4; ++m) {
      int off = (64 * wm + 16 * m + l15) * LDS_ROW + (k0 + aK) * 2;
      aH[m].u[0] = *(const uint4*)(smem + LDS_AHI + off);
      aH[m].u[1] = *(const uint4*)(smem + LDS_AHI + off + 32);
      aL[m].u[0] = *(const uint4*)(smem + LDS_ALO + off);
      aL[m].u[1] = *(const uint4*)(smem + LDS_ALO + off + 32);
    }
    #pragma unroll
    for (int n = 0; n < 2; ++n) {
      int boff = (32 * wn + 16 * n + l15) * LDS_ROW + (k0 + bK) * 2;
      BfOp bH, bL;
      bH.u[0] = *(const uint4*)(smem + LDS_BHI + boff);
      bH.u[1] = *(const uint4*)(smem + LDS_BHI + boff + 32);
      bL.u[0] = *(const uint4*)(smem + LDS_BLO + boff);
      bL.u[1] = *(const uint4*)(smem + LDS_BLO + boff + 32);
      #pragma unroll
      for (int m = 0; m < 4; ++m) {
        acc[m][n] = __builtin_amdgcn_wmma_f32_16x16x32_bf16(
            false, aH[m].v, false, bH.v, (short)0, acc[m][n], false, false);
        acc[m][n] = __builtin_amdgcn_wmma_f32_16x16x32_bf16(
            false, aH[m].v, false, bL.v, (short)0, acc[m][n], false, false);
        acc[m][n] = __builtin_amdgcn_wmma_f32_16x16x32_bf16(
            false, aL[m].v, false, bH.v, (short)0, acc[m][n], false, false);
      }
    }
  }

  #pragma unroll
  for (int m = 0; m < 4; ++m)
    #pragma unroll
    for (int n = 0; n < 2; ++n) {
      int col = nBase + 32 * wn + 16 * n + l15;
      #pragma unroll
      for (int r = 0; r < 8; ++r) {
        int row = mBase + 64 * wm + 16 * m + r + 8 * hs;
        float v = acc[m][n][r];
        // sigmoid(v) >= 0.5  <=>  v >= 0 ; strict upper triangle only
        float o = (row < col && v >= 0.0f) ? (1.0f / (1.0f + __expf(-v))) : 0.0f;
        out[(size_t)row * N_NODES + col] = o;
      }
    }
}

// ---------------- Edge scatter: out[src,dst] = 1.0 -------------------------
__global__ __launch_bounds__(256) void edge_set_kernel(
    const long long* __restrict__ ei, float* __restrict__ out, int E)
{
  int i = blockIdx.x * 256 + threadIdx.x;
  if (i < E) {
    long long s = ei[i];
    long long d = ei[E + i];
    out[(size_t)s * N_NODES + (size_t)d] = 1.0f;
  }
}

// ---------------------------------------------------------------------------
extern "C" void kernel_launch(void* const* d_in, const int* in_sizes, int n_in,
                              void* d_out, int out_size, void* d_ws, size_t ws_size,
                              hipStream_t stream)
{
  (void)n_in; (void)out_size; (void)ws_size;

  const float*     x     = (const float*)d_in[0];
  const long long* ei    = (const long long*)d_in[1];   // int64 per reference
  const float*     W1    = (const float*)d_in[2];
  const float*     b1    = (const float*)d_in[3];
  const float*     gamma = (const float*)d_in[4];
  const float*     beta  = (const float*)d_in[5];
  const float*     W2    = (const float*)d_in[6];
  const float*     b2    = (const float*)d_in[7];
  float*           out   = (float*)d_out;

  // workspace layout (all 256B aligned)
  char* ws = (char*)d_ws;
  float*          h       = (float*)(ws);                        // 16 MB
  float*          partial = (float*)(ws + 16777216);             // 256 KB
  float*          scale   = (float*)(ws + 17039360);             // 2 KB
  float*          shiftv  = (float*)(ws + 17041408);             // 2 KB
  unsigned short* ehi     = (unsigned short*)(ws + 17043456);    // 2 MB
  unsigned short* elo     = (unsigned short*)(ws + 19140608);    // 2 MB

  // 1) h = x @ W1^T + b1            (16384 waves)
  gemm1_kernel<<<2048, 256, 0, stream>>>(x, W1, b1, h);
  // 2) BatchNorm statistics
  bn_partial_kernel<<<64, 512, 0, stream>>>(h, partial);
  bn_finalize_kernel<<<1, 512, 0, stream>>>(partial, gamma, beta, scale, shiftv);
  // 3) normalize + ReLU in place
  bn_apply_kernel<<<(N_NODES * HID) / 256, 256, 0, stream>>>(h, scale, shiftv);
  // 4) emb = hn @ W2^T + b2 -> bf16 hi/lo  (4096 waves)
  gemm2_kernel<<<512, 256, 0, stream>>>(h, W2, b2, ehi, elo);
  // 5) masked sigmoid(emb @ emb^T): 64x64 blocks of 128x128, TDM-staged LDS
  syrk_sigmoid_kernel<<<4096, 256, (size_t)LDS_TOTAL, stream>>>(ehi, elo, out);
  // 6) scatter original edges as 1.0
  int E = in_sizes[1] / 2;
  edge_set_kernel<<<(E + 255) / 256, 256, 0, stream>>>(ei, out, E);
}
